// SwinTransformerBlock_35115652612544
// MI455X (gfx1250) — matrix operand order repaired and checked by reading, not compiled, not run on hardware
//
#include <hip/hip_runtime.h>
#include <hip/hip_bf16.h>
#include <math.h>

// ---------------- Swin constants ----------------
#define Bb     32
#define Hh     56
#define Ww     56
#define Cc     192
#define WS     7
#define SHIFT  3
#define HEADS  6
#define Dd     32            // C / HEADS
#define HID    768
#define Nn     49            // WS*WS
#define NW     64            // (H/WS)*(W/WS)
#define NWIN   (Bb * NW)     // 2048
#define TOK    (Bb * Hh * Ww) // 100352 (== NWIN*Nn)

typedef unsigned short u16;
typedef unsigned int   u32;
typedef unsigned long long u64;
typedef __attribute__((ext_vector_type(16))) __bf16 v16bf;
typedef __attribute__((ext_vector_type(8)))  float  v8f;
typedef __attribute__((ext_vector_type(8)))  u32    v8u;

// ---------------- bf16 helpers (manual RNE, no dependence on __bf16 arith) ----
__device__ __forceinline__ u16 f2bf(float f) {
  u32 u = __builtin_bit_cast(u32, f);
  u32 r = u + 0x7FFFu + ((u >> 16) & 1u);
  return (u16)(r >> 16);
}
__device__ __forceinline__ float bf2f(u16 h) {
  u32 u = ((u32)h) << 16;
  return __builtin_bit_cast(float, u);
}

// ---------------- CDNA5 async copy: global -> LDS (ASYNCcnt tracked) --------
// VDST = per-lane LDS byte offset (low 32 bits of generic shared pointer),
// VADDR = 64-bit global address (GV mode, saddr = off).
__device__ __forceinline__ void async_load_b128(unsigned lds_off, u64 gaddr) {
  asm volatile("global_load_async_to_lds_b128 %0, %1, off"
               :: "v"(lds_off), "v"(gaddr)
               : "memory");
}
__device__ __forceinline__ void wait_asynccnt0() {
  asm volatile("s_wait_asynccnt 0" ::: "memory");
}

// ---------------- WMMA fragment loaders (wave32, contiguous-k order for A and B) --------
// A 16x32 bf16: lane L: row = L&15, half h = L>>4.
//   v16 element e -> K = 16h + e
// Read as 8 dword pairs (K even-aligned) from row-major LDS (stride in u16).
__device__ __forceinline__ v16bf frag_A(const u16* base, int strideU16, int lane) {
  int r = lane & 15, hh = lane >> 4;
  const u32* p = (const u32*)(base + (size_t)r * strideU16);
  int o = hh * 8;
  v8u t;
  t[0] = p[o + 0]; t[1] = p[o + 1]; t[2] = p[o + 2]; t[3] = p[o + 3];
  t[4] = p[o + 4]; t[5] = p[o + 5]; t[6] = p[o + 6]; t[7] = p[o + 7];
  return __builtin_bit_cast(v16bf, t);
}
// B 32x16 bf16: lane L: col = L&15, half h = L>>4; element e -> K = 16h + e.
// Requires B stored K-contiguous per column (i.e. transposed): baseT[col][k].
__device__ __forceinline__ v16bf frag_B(const u16* baseT, int colStrideU16, int lane) {
  int c = lane & 15, hh = lane >> 4;
  const u32* p = (const u32*)(baseT + (size_t)c * colStrideU16);
  int o = hh * 8;
  v8u t;
#pragma unroll
  for (int j = 0; j < 8; ++j) t[j] = p[o + j];
  return __builtin_bit_cast(v16bf, t);
}
__device__ __forceinline__ v8f wmma_bf16(v16bf a, v16bf b, v8f c) {
  return __builtin_amdgcn_wmma_f32_16x16x32_bf16(false, a, false, b, (short)0, c,
                                                 false, false);
}

// ---------------- fp32 -> bf16 weight conversion ----------------------------
__global__ void cvt_bf16_kernel(const float* __restrict__ in, u16* __restrict__ out, int n) {
  int i = blockIdx.x * 256 + threadIdx.x;
  if (i < n) out[i] = f2bf(in[i]);
}

// ---------------- LayerNorm (optionally fused shift + window partition) -----
// One 64-thread block per output row; 3 channels per thread over C=192.
__launch_bounds__(64)
__global__ void ln_kernel(const float* __restrict__ x, const float* __restrict__ w,
                          const float* __restrict__ bia, u16* __restrict__ out,
                          int windowed) {
  int row = blockIdx.x;
  size_t src;
  if (windowed) {
    int win = row / Nn, n = row % Nn;
    int b_ = win >> 6, wi = win & 63;
    int wh = wi >> 3, wwi = wi & 7;
    int ii = n / WS, jj = n % WS;
    int y = (wh * WS + ii + SHIFT) % Hh;   // roll(-SHIFT)
    int z = (wwi * WS + jj + SHIFT) % Ww;
    src = (size_t)b_ * (Hh * Ww) + (size_t)y * Ww + z;
  } else {
    src = (size_t)row;
  }
  const float* xp = x + src * Cc;
  int t = threadIdx.x;
  float v0 = xp[t], v1 = xp[t + 64], v2 = xp[t + 128];
  __shared__ float red[64];
  red[t] = v0 + v1 + v2;
  __syncthreads();
  for (int o = 32; o > 0; o >>= 1) { if (t < o) red[t] += red[t + o]; __syncthreads(); }
  float mu = red[0] * (1.0f / Cc);
  __syncthreads();
  float d0 = v0 - mu, d1 = v1 - mu, d2 = v2 - mu;
  red[t] = d0 * d0 + d1 * d1 + d2 * d2;
  __syncthreads();
  for (int o = 32; o > 0; o >>= 1) { if (t < o) red[t] += red[t + o]; __syncthreads(); }
  float rs = rsqrtf(red[0] * (1.0f / Cc) + 1e-5f);
  u16* op = out + (size_t)row * Cc;
  op[t]       = f2bf(d0 * rs * w[t]       + bia[t]);
  op[t + 64]  = f2bf(d1 * rs * w[t + 64]  + bia[t + 64]);
  op[t + 128] = f2bf(d2 * rs * w[t + 128] + bia[t + 128]);
}

// ---------------- Tiled bf16 WMMA GEMM with fused epilogues ------------------
// C[M,Nc] = A[M,K] @ Bw[K,Nc]; block tile 64x64, 4 waves, K-step 32.
// A tile: async DMA global->LDS (ASYNCcnt).  B tile: load + transpose via LDS.
// mode 0: out_bf = acc + bias                      (QKV)
// mode 1: window-reverse+roll scatter, outf = resid + acc + bias   (proj)
// mode 2: out_bf = gelu_exact(acc + bias)          (fc1)
// mode 3: outf = resid + acc + bias, nontemporal   (fc2 -> final output)
__launch_bounds__(128)
__global__ void gemm_bf16_kernel(const u16* __restrict__ A, const u16* __restrict__ Bw,
                                 const float* __restrict__ bias,
                                 int M, int Nc, int K, int mode,
                                 u16* __restrict__ outb, float* __restrict__ outf,
                                 const float* __restrict__ resid) {
  __shared__ __align__(16) u16 As[64 * 32];   // [row][k]
  __shared__ u16 Bt[64 * 32];                 // [col][k] (transposed tile)
  int t = threadIdx.x, lane = t & 31, wv = t >> 5;
  int row0 = blockIdx.x * 64, col0 = blockIdx.y * 64;

  v8f acc[4] = {};
  for (int k0 = 0; k0 < K; k0 += 32) {
    // A tile: 64x32 bf16 = 256 chunks of 16B, 2 per thread, async DMA to LDS.
#pragma unroll
    for (int i = 0; i < 2; ++i) {
      int q = i * 128 + t;                 // chunk id
      int r = q >> 2, cp = q & 3;          // row, 8-element part
      const u16* gp = A + (size_t)(row0 + r) * K + k0 + cp * 8;
      unsigned lds = (unsigned)(u64)(const void*)(As + r * 32 + cp * 8);
      async_load_b128(lds, (u64)(const void*)gp);
      if (k0 + 32 < K)                     // prefetch next K-step chunk
        __builtin_prefetch(gp + 32, 0, 1);
    }
    // B tile: 32x64 bf16, store transposed [col][k]
#pragma unroll
    for (int i = 0; i < 8; ++i) {
      int u = i * 128 + t;
      int kr = u >> 5, cp = u & 31;
      u32 d = *(const u32*)(Bw + (size_t)(k0 + kr) * Nc + col0 + cp * 2);
      Bt[(cp * 2 + 0) * 32 + kr] = (u16)(d & 0xFFFFu);
      Bt[(cp * 2 + 1) * 32 + kr] = (u16)(d >> 16);
    }
    wait_asynccnt0();
    __syncthreads();
    v16bf a = frag_A(As + wv * 16 * 32, 32, lane);
#pragma unroll
    for (int cn = 0; cn < 4; ++cn) {
      v16bf bfr = frag_B(Bt + cn * 16 * 32, 32, lane);
      acc[cn] = wmma_bf16(a, bfr, acc[cn]);
    }
    __syncthreads();
  }

  // Epilogue. 16x16 f32 C layout: element i -> row = i + 8*(lane>>4), col = lane&15.
  int cl = lane & 15, hh = lane >> 4;
#pragma unroll
  for (int cn = 0; cn < 4; ++cn) {
    int colg = col0 + cn * 16 + cl;
    float bv = bias[colg];
#pragma unroll
    for (int i = 0; i < 8; ++i) {
      int rg = row0 + wv * 16 + i + hh * 8;
      float v = acc[cn][i] + bv;
      if (mode == 0) {
        outb[(size_t)rg * Nc + colg] = f2bf(v);
      } else if (mode == 2) {
        float g = 0.5f * v * (1.0f + erff(v * 0.70710678118654752f));
        outb[(size_t)rg * Nc + colg] = f2bf(g);
      } else if (mode == 3) {
        size_t o = (size_t)rg * Nc + colg;
        __builtin_nontemporal_store(resid[o] + v, &outf[o]);
      } else { // mode 1: proj + residual + window reverse + roll(+SHIFT)
        int win = rg / Nn, n = rg % Nn;
        int b_ = win >> 6, wi = win & 63;
        int wh = wi >> 3, wwi = wi & 7;
        int ii = n / WS, jj = n % WS;
        int p = wh * WS + ii, q = wwi * WS + jj;
        int y = (p + SHIFT) % Hh, z = (q + SHIFT) % Ww;
        size_t o = ((size_t)b_ * (Hh * Ww) + (size_t)y * Ww + z) * Cc + colg;
        outf[o] = resid[o] + v;
      }
    }
  }
}

// ---------------- Windowed attention: one block per (window, head) ----------
// S = (q*scale) @ k^T + bias + mask; P = softmax(S); O = P @ v.
// N=49 padded to 64 rows (zero pad). d=32 == one bf16 WMMA K step.
__launch_bounds__(128)
__global__ void attn_kernel(const u16* __restrict__ qkv,
                            const float* __restrict__ bias_table,
                            const float* __restrict__ attn_mask,
                            const int* __restrict__ rel_index,
                            u16* __restrict__ owin) {
  // smem carving: qs 4KB | ks 4KB | vsT 4KB | bm 9616B | sp 16KB  (~38KB)
  __shared__ __align__(16) char smem[4096 * 3 + 9616 + 64 * 64 * 4];
  u16*   qs  = (u16*)smem;
  u16*   ks  = (u16*)(smem + 4096);
  u16*   vsT = (u16*)(smem + 8192);          // [d=32][m=64] transposed
  float* bm  = (float*)(smem + 12288);       // [49*49]
  float* sp  = (float*)(smem + 12288 + 9616);// [64][64]
  u16*   ps  = (u16*)smem;                   // overlays qs+ks after S phase

  int blk = blockIdx.x;
  int w = blk / HEADS, h = blk % HEADS;
  int wi = w & 63;                            // window index within image
  int t = threadIdx.x, lane = t & 31, wv = t >> 5;
  const float scale = 0.17677669529663687f;   // 32^-0.5

  size_t qbase = (size_t)w * Nn * (3 * Cc) + (size_t)h * Dd;
#pragma unroll
  for (int i = 0; i < 16; ++i) {             // 64x32 padded tiles
    int e = i * 128 + t;
    int r = e >> 5, c = e & 31;
    u16 qv = 0, kv = 0, vv = 0;
    if (r < Nn) {
      size_t rb = qbase + (size_t)r * (3 * Cc);
      qv = f2bf(bf2f(qkv[rb + c]) * scale);
      kv = qkv[rb + Cc + c];
      vv = qkv[rb + 2 * Cc + c];
    }
    qs[r * 32 + c] = qv;
    ks[r * 32 + c] = kv;
    vsT[c * 64 + r] = vv;
  }
  for (int idx = t; idx < Nn * Nn; idx += 128) {
    bm[idx] = bias_table[rel_index[idx] * HEADS + h] +
              attn_mask[(size_t)wi * (Nn * Nn) + idx];
  }
  __syncthreads();

  // ---- S = q @ k^T : wave wv handles 16-row slab; K=d=32 -> 1 WMMA per tile
  {
    int cl = lane & 15, hh = lane >> 4;
    v16bf a = frag_A(qs + wv * 16 * 32, 32, lane);
#pragma unroll
    for (int cn = 0; cn < 4; ++cn) {
      // B = k^T (32x16): b[e] = k[col][K] -> ks row-major is already K-contiguous
      v16bf bfr = frag_B(ks + cn * 16 * 32, 32, lane);
      v8f c = {};
      c = wmma_bf16(a, bfr, c);
#pragma unroll
      for (int i = 0; i < 8; ++i)
        sp[(wv * 16 + i + hh * 8) * 64 + cn * 16 + cl] = c[i];
    }
  }
  __syncthreads();

  // ---- softmax rows (thread r handles row r); write P bf16 into ps overlay
  if (t < 64) {
    int r = t;
    if (r < Nn) {
      float mx = -1e30f;
      for (int m = 0; m < Nn; ++m)
        mx = fmaxf(mx, sp[r * 64 + m] + bm[r * Nn + m]);
      float s = 0.0f;
      for (int m = 0; m < Nn; ++m)
        s += __expf(sp[r * 64 + m] + bm[r * Nn + m] - mx);
      float inv = 1.0f / s;
      for (int m = 0; m < Nn; ++m)
        ps[r * 64 + m] = f2bf(__expf(sp[r * 64 + m] + bm[r * Nn + m] - mx) * inv);
      for (int m = Nn; m < 64; ++m) ps[r * 64 + m] = 0;
    } else {
      for (int m = 0; m < 64; ++m) ps[r * 64 + m] = 0;
    }
  }
  __syncthreads();

  // ---- O = P @ v : K = 64 (2 chunks of 32), 2 column tiles of d
  {
    int cl = lane & 15, hh = lane >> 4;
    v16bf a0 = frag_A(ps + wv * 16 * 64, 64, lane);        // K chunk 0
    v16bf a1 = frag_A(ps + wv * 16 * 64 + 32, 64, lane);   // K chunk 1
#pragma unroll
    for (int ct = 0; ct < 2; ++ct) {
      v16bf b0 = frag_B(vsT + ct * 16 * 64, 64, lane);       // K rows 0..31
      v16bf b1 = frag_B(vsT + ct * 16 * 64 + 32, 64, lane);  // K rows 32..63
      v8f c = {};
      c = wmma_bf16(a0, b0, c);
      c = wmma_bf16(a1, b1, c);
#pragma unroll
      for (int i = 0; i < 8; ++i) {
        int rl = wv * 16 + i + hh * 8;
        if (rl < Nn)
          owin[((size_t)w * Nn + rl) * Cc + h * Dd + ct * 16 + cl] = f2bf(c[i]);
      }
    }
  }
}

// ---------------- host launcher ---------------------------------------------
extern "C" void kernel_launch(void* const* d_in, const int* in_sizes, int n_in,
                              void* d_out, int out_size, void* d_ws, size_t ws_size,
                              hipStream_t stream) {
  (void)in_sizes; (void)n_in; (void)out_size; (void)ws_size;
  const float* x          = (const float*)d_in[0];
  const float* norm1_w    = (const float*)d_in[1];
  const float* norm1_b    = (const float*)d_in[2];
  const float* qkv_w      = (const float*)d_in[3];
  const float* qkv_b      = (const float*)d_in[4];
  const float* bias_table = (const float*)d_in[5];
  const float* proj_w     = (const float*)d_in[6];
  const float* proj_b     = (const float*)d_in[7];
  const float* norm2_w    = (const float*)d_in[8];
  const float* norm2_b    = (const float*)d_in[9];
  const float* fc1_w      = (const float*)d_in[10];
  const float* fc1_b      = (const float*)d_in[11];
  const float* fc2_w      = (const float*)d_in[12];
  const float* fc2_b      = (const float*)d_in[13];
  const float* attn_mask  = (const float*)d_in[14];
  const int*   rel_index  = (const int*)d_in[15];
  float* out = (float*)d_out;

  char* ws = (char*)d_ws;
  size_t off = 0;
  auto carve = [&](size_t bytes) {
    void* p = ws + off;
    off += (bytes + 255) & ~(size_t)255;
    return p;
  };
  u16*   hwin = (u16*)carve((size_t)TOK * Cc * 2);        // LN1 windows / LN2 out (reused)
  u16*   qkvb = (u16*)carve((size_t)TOK * 3 * Cc * 2);    // QKV
  u16*   owin = (u16*)carve((size_t)TOK * Cc * 2);        // attention out (window layout)
  float* x1   = (float*)carve((size_t)TOK * Cc * 4);      // shortcut + proj
  u16*   hid  = (u16*)carve((size_t)TOK * HID * 2);       // gelu(fc1)
  u16*   wqkv = (u16*)carve((size_t)Cc * 3 * Cc * 2);
  u16*   wprj = (u16*)carve((size_t)Cc * Cc * 2);
  u16*   wfc1 = (u16*)carve((size_t)Cc * HID * 2);
  u16*   wfc2 = (u16*)carve((size_t)HID * Cc * 2);

  // weights -> bf16
  cvt_bf16_kernel<<<(Cc * 3 * Cc + 255) / 256, 256, 0, stream>>>(qkv_w, wqkv, Cc * 3 * Cc);
  cvt_bf16_kernel<<<(Cc * Cc + 255) / 256, 256, 0, stream>>>(proj_w, wprj, Cc * Cc);
  cvt_bf16_kernel<<<(Cc * HID + 255) / 256, 256, 0, stream>>>(fc1_w, wfc1, Cc * HID);
  cvt_bf16_kernel<<<(HID * Cc + 255) / 256, 256, 0, stream>>>(fc2_w, wfc2, HID * Cc);

  // LN1 + shift + window partition
  ln_kernel<<<TOK, 64, 0, stream>>>(x, norm1_w, norm1_b, hwin, 1);

  // QKV GEMM: [TOK,192] @ [192,576]
  gemm_bf16_kernel<<<dim3(TOK / 64, (3 * Cc) / 64), 128, 0, stream>>>(
      hwin, wqkv, qkv_b, TOK, 3 * Cc, Cc, /*mode=*/0, qkvb, nullptr, nullptr);

  // attention per (window, head)
  attn_kernel<<<NWIN * HEADS, 128, 0, stream>>>(qkvb, bias_table, attn_mask,
                                                rel_index, owin);

  // proj GEMM + residual + window reverse/roll scatter -> x1
  gemm_bf16_kernel<<<dim3(TOK / 64, Cc / 64), 128, 0, stream>>>(
      owin, wprj, proj_b, TOK, Cc, Cc, /*mode=*/1, nullptr, x1, x);

  // LN2 -> bf16 (reuse hwin)
  ln_kernel<<<TOK, 64, 0, stream>>>(x1, norm2_w, norm2_b, hwin, 0);

  // fc1 GEMM + exact GELU: [TOK,192] @ [192,768]
  gemm_bf16_kernel<<<dim3(TOK / 64, HID / 64), 128, 0, stream>>>(
      hwin, wfc1, fc1_b, TOK, HID, Cc, /*mode=*/2, hid, nullptr, nullptr);

  // fc2 GEMM + residual -> output: [TOK,768] @ [768,192]
  gemm_bf16_kernel<<<dim3(TOK / 64, Cc / 64), 128, 0, stream>>>(
      hid, wfc2, fc2_b, TOK, Cc, HID, /*mode=*/3, nullptr, out, x1);
}
